// LocalNet_15247133901595
// MI455X (gfx1250) — compile-verified, matrix operand order checked by reading, not run on hardware
//
#include <hip/hip_runtime.h>

typedef __attribute__((ext_vector_type(16))) __bf16 v16bf;
typedef __attribute__((ext_vector_type(8)))  float  v8f;
typedef unsigned int u32;
typedef int v4i __attribute__((vector_size(4 * sizeof(int))));

#if __has_builtin(__builtin_amdgcn_global_load_async_to_lds_b128)
#define HAVE_ASYNC_LDS 1
#else
#define HAVE_ASYNC_LDS 0
#endif

// ---------- bf16 <-> f32 helpers (bit ops only; no reliance on __bf16 arith) ----
__device__ __forceinline__ __bf16 f2bf(float f) {
  union { float f; u32 u; } a; a.f = f;
  u32 r = a.u + 0x7fffu + ((a.u >> 16) & 1u);         // round-to-nearest-even
  union { unsigned short s; __bf16 b; } o; o.s = (unsigned short)(r >> 16);
  return o.b;
}
__device__ __forceinline__ float bf2f(__bf16 b) {
  union { unsigned short s; __bf16 b; } i; i.b = b;
  union { float f; u32 u; } o; o.u = (u32)i.s << 16;
  return o.f;
}
__device__ __forceinline__ void store_out(__bf16* p, float v) { *p = f2bf(v); }
__device__ __forceinline__ void store_out(float* p, float v)  { *p = v; }

// ---------- Async global -> LDS copy (CDNA5), with synchronous fallback ---------
// Builtin's strict param types (from compiler diagnostic): global v4i*, LDS v4i*.
__device__ __forceinline__ void cp_async_b128(const void* g, void* l) {
#if HAVE_ASYNC_LDS
  typedef __attribute__((address_space(1))) v4i GV;
  typedef __attribute__((address_space(3))) v4i LV;
  __builtin_amdgcn_global_load_async_to_lds_b128((GV*)(void*)g, (LV*)l, 0, 0);
#else
  *reinterpret_cast<uint4*>(l) = *reinterpret_cast<const uint4*>(g);
#endif
}
__device__ __forceinline__ void async_wait_all() {
#if HAVE_ASYNC_LDS
#if __has_builtin(__builtin_amdgcn_s_wait_asynccnt)
  __builtin_amdgcn_s_wait_asynccnt(0);
#else
  asm volatile("s_wait_asynccnt 0x0" ::: "memory");
#endif
#endif
}

// ---------- WMMA core -----------------------------------------------------------
__device__ __forceinline__ v8f wmma_acc(v16bf a, v16bf b, v8f c) {
  // D = A(16x32 bf16) * B(32x16 bf16) + C(f32)
  return __builtin_amdgcn_wmma_f32_16x16x32_bf16(
      /*neg_a=*/false, a, /*neg_b=*/false, b,
      /*c_mod=*/(short)0, c, /*reuse_a=*/false, /*reuse_b=*/false);
}

// Load one 16x32 bf16 fragment from an LDS tile stored row-major
// (row = M for A, row = N for B; 32 K values contiguous per row, row stride ldr).
// ISA 16-bit operand layout: lane half h = lane>>4, row = lane&15;
//   elements 0..7  -> K = h*8 + 0..7
//   elements 8..15 -> K = 16 + h*8 + 0..7   (pairs are K-consecutive -> b32 reads)
__device__ __forceinline__ v16bf load_frag(const __bf16* base, int ldr, int lane) {
  const int r = lane & 15;
  const int h = lane >> 4;
  const __bf16* p = base + r * ldr + h * 8;
  v16bf f;
  u32* fw = reinterpret_cast<u32*>(&f);
#pragma unroll
  for (int j = 0; j < 4; ++j) {
    fw[j]     = *reinterpret_cast<const u32*>(p + 2 * j);
    fw[j + 4] = *reinterpret_cast<const u32*>(p + 16 + 2 * j);
  }
  return f;
}

// ---------- Generic batched GEMM: C[z] = relu(A[z] * B[z]^T + bias[z]) ----------
// A: [z][M][K] bf16 row-major (row stride lda), B: [z][N][K] bf16 (row stride K),
// C element (m,n) stored at Cz + m*ldcm + n*ldcn, Cz = C + z*strideCz.
template <int BM, int BN, bool RELU, typename OT>
__global__ __launch_bounds__(256) void gemm_bias_act(
    const __bf16* __restrict__ A, long long strideAz, int lda,
    const __bf16* __restrict__ B, long long strideBz,
    const float*  __restrict__ bias, int strideBiasZ,
    OT* __restrict__ C, long long strideCz, long long ldcm, long long ldcn,
    int K) {
  static_assert(BM == 64, "8 wave32s arranged 4(M) x 2(N)");
  constexpr int BK  = 32;
  constexpr int LDR = BK + 8;       // +8 bf16 pad: kills LDS bank conflicts
  constexpr int HN  = BN / 2;       // per-wave N extent
  constexpr int NF  = HN / 16;      // N fragments per wave

  __shared__ __bf16 As[2][BM * LDR];
  __shared__ __bf16 Bs[2][BN * LDR];

  const int tid  = threadIdx.x;
  const int lane = tid & 31;
  const int wid  = tid >> 5;
  const int wm   = wid >> 1;        // 0..3
  const int wn   = wid & 1;         // 0..1
  const int z    = blockIdx.z;
  const long long m0 = (long long)blockIdx.x * BM;
  const long long n0 = (long long)blockIdx.y * BN;

  const __bf16* Ag = A + (long long)z * strideAz + m0 * lda;
  const __bf16* Bg = B + (long long)z * strideBz + n0 * (long long)K;

  auto stage = [&](int buf, int k0) {
#pragma unroll 2
    for (int c = tid; c < BM * 4; c += 256) {          // 4 x 16B chunks per row
      const int row = c >> 2, kc = c & 3;
      cp_async_b128(Ag + row * lda + k0 + kc * 8,
                    &As[buf][row * LDR + kc * 8]);
    }
#pragma unroll 2
    for (int c = tid; c < BN * 4; c += 256) {
      const int row = c >> 2, kc = c & 3;
      cp_async_b128(Bg + row * K + k0 + kc * 8,
                    &Bs[buf][row * LDR + kc * 8]);
    }
  };

  v8f zero = {};
  v8f acc[NF];
#pragma unroll
  for (int f = 0; f < NF; ++f) acc[f] = zero;

  const int nsteps = K / BK;
  stage(0, 0);
  async_wait_all();
  __syncthreads();
  for (int s = 0; s < nsteps; ++s) {
    const int buf = s & 1;
    if (s + 1 < nsteps) stage(buf ^ 1, (s + 1) * BK);
    const v16bf af = load_frag(&As[buf][(wm * 16) * LDR], LDR, lane);
#pragma unroll
    for (int f = 0; f < NF; ++f) {
      const v16bf bfr = load_frag(&Bs[buf][(wn * HN + f * 16) * LDR], LDR, lane);
      acc[f] = wmma_acc(af, bfr, acc[f]);
    }
    async_wait_all();
    __syncthreads();
  }

  // Epilogue. C/D layout: lane half hh = lane>>4, n = lane&15, element e -> M = hh*8+e.
  const float* bz = bias + (long long)z * strideBiasZ;
  OT* Cz = C + (long long)z * strideCz;
  const int r  = lane & 15;
  const int hh = lane >> 4;
#pragma unroll
  for (int f = 0; f < NF; ++f) {
    const int ncol = (int)n0 + wn * HN + f * 16 + r;
    const float bv = bz[ncol];
#pragma unroll
    for (int e = 0; e < 8; ++e) {
      const long long mrow = m0 + wm * 16 + hh * 8 + e;
      float v = acc[f][e] + bv;
      if (RELU) v = fmaxf(v, 0.0f);
      store_out(&Cz[mrow * ldcm + (long long)ncol * ldcn], v);
    }
  }
}

// ---------- Layer-1 GEMM with fused im2col gather of A from H0 ------------------
// Per position q (blockIdx.z): out(b, o) = relu( sum_k patch(b,k) * W1b(q,o,k) + b1(q,o) )
// patch(b, k=i*16+ky*4+kx) = H0[b][i][qy*2+ky][qx*2+kx], K = 1024.
// Output written straight into FC-flatten layout: H1[b*25088 + o*196 + q].
__global__ __launch_bounds__(256) void gemm_local1(
    const __bf16* __restrict__ H0,   // (256,64,31,31) bf16
    const __bf16* __restrict__ B,    // (196,128,1024) bf16, [q][o][k]
    const float*  __restrict__ bias, // (196,128)
    __bf16* __restrict__ H1) {       // (256,25088)
  constexpr int BM = 64, BN = 128, BK = 32, LDR = BK + 8, HN = 64, NF = 4, K = 1024;
  __shared__ __bf16 As[2][BM * LDR];
  __shared__ __bf16 Bs[2][BN * LDR];

  const int tid  = threadIdx.x;
  const int lane = tid & 31;
  const int wid  = tid >> 5;
  const int wm   = wid >> 1;
  const int wn   = wid & 1;
  const int q    = blockIdx.z;
  const int qy   = q / 14, qx = q % 14;
  const int m0   = blockIdx.x * BM;

  const __bf16* Bg = B + (long long)q * 128 * 1024;

  auto stage = [&](int buf, int k0) {
    // A: gather 64 rows x 32 K straight from the activation map (fused im2col).
    // All indices fit in 32 bits (H0 has 15.7M elements) -> int math only.
#pragma unroll 8
    for (int t = tid; t < BM * BK; t += 256) {
      const int row = t >> 5, kk = t & 31;
      const int k = k0 + kk;
      const int i = k >> 4, rr = (k >> 2) & 3, cc = k & 3;
      const int b = m0 + row;
      As[buf][row * LDR + kk] =
          H0[((b * 64 + i) * 31 + (qy * 2 + rr)) * 31 + (qx * 2 + cc)];
    }
#pragma unroll 2
    for (int c = tid; c < BN * 4; c += 256) {
      const int row = c >> 2, kc = c & 3;
      cp_async_b128(Bg + row * K + k0 + kc * 8,
                    &Bs[buf][row * LDR + kc * 8]);
    }
  };

  v8f zero = {};
  v8f acc[NF];
#pragma unroll
  for (int f = 0; f < NF; ++f) acc[f] = zero;

  stage(0, 0);
  async_wait_all();
  __syncthreads();
  for (int s = 0; s < K / BK; ++s) {
    const int buf = s & 1;
    if (s + 1 < K / BK) stage(buf ^ 1, (s + 1) * BK);
    const v16bf af = load_frag(&As[buf][(wm * 16) * LDR], LDR, lane);
#pragma unroll
    for (int f = 0; f < NF; ++f) {
      const v16bf bfr = load_frag(&Bs[buf][(wn * HN + f * 16) * LDR], LDR, lane);
      acc[f] = wmma_acc(af, bfr, acc[f]);
    }
    async_wait_all();
    __syncthreads();
  }

  const float* bz = bias + q * 128;
  const int r  = lane & 15;
  const int hh = lane >> 4;
#pragma unroll
  for (int f = 0; f < NF; ++f) {
    const int ncol = wn * HN + f * 16 + r;
    const float bv = bz[ncol];
#pragma unroll
    for (int e = 0; e < 8; ++e) {
      const int mrow = m0 + wm * 16 + hh * 8 + e;
      // H1 has 6.4M elements -> int index math is safe.
      H1[mrow * 25088 + ncol * 196 + q] = f2bf(fmaxf(acc[f][e] + bv, 0.0f));
    }
  }
}

// ---------- Prep / conversion kernels -------------------------------------------
// W0 (961,64,3,4,4) fp32 -> B0 (961,64,64) bf16 with K 48->64 zero pad.
__global__ void k_cvt_w0(const float* __restrict__ w, __bf16* __restrict__ out) {
  const int i = blockIdx.x * 256 + threadIdx.x;        // 961*64*64
  if (i >= 961 * 64 * 64) return;
  const int k = i & 63, po = i >> 6;
  out[i] = (k < 48) ? f2bf(w[po * 48 + k]) : f2bf(0.0f);
}

// x (256,3,64,64) fp32 -> P0 (961,256,64) bf16 im2col (K = i*16+ky*4+kx, pad 48->64).
__global__ void k_im2col0(const float* __restrict__ x, __bf16* __restrict__ p0) {
  const long long i = (long long)blockIdx.x * 256 + threadIdx.x;  // 961*256*64
  if (i >= 961LL * 256 * 64) return;
  const int k = (int)(i & 63);
  const int b = (int)((i >> 6) & 255);
  const int p = (int)(i >> 14);
  const int py = p / 31, px = p % 31;
  __bf16 v = f2bf(0.0f);
  if (k < 48) {
    const int ch = k >> 4, rr = (k >> 2) & 3, cc = k & 3;
    v = f2bf(x[((b * 3 + ch) * 64 + (py * 2 + rr)) * 64 + (px * 2 + cc)]);
  }
  p0[i] = v;
}

// Plain fp32 -> bf16 (W1 keeps its natural [q][o][k] order).
__global__ void k_cvt(const float* __restrict__ in, __bf16* __restrict__ out, long long n) {
  const long long i = (long long)blockIdx.x * 256 + threadIdx.x;
  if (i < n) out[i] = f2bf(in[i]);
}

// fcW0 (25088,1024) fp32 -> W2 (1024,25088) bf16 transposed to [n][k].
__global__ void k_cvt_fcw0T(const float* __restrict__ w, __bf16* __restrict__ out) {
  const long long i = (long long)blockIdx.x * 256 + threadIdx.x;  // 1024*25088
  if (i >= 1024LL * 25088) return;
  const int n = (int)(i / 25088);
  const int k = (int)(i % 25088);
  out[i] = f2bf(w[(long long)k * 1024 + n]);
}

// FC1: (256,1024) x (1024,10) + bias -> fp32 out. N=10 -> scalar kernel.
__global__ void k_fc1(const __bf16* __restrict__ h2, const float* __restrict__ w,
                      const float* __restrict__ b, float* __restrict__ out) {
  const int t = blockIdx.x * 256 + threadIdx.x;
  if (t >= 2560) return;
  const int bi = t / 10, o = t % 10;
  const __bf16* hr = h2 + bi * 1024;
  float s = b[o];
  for (int k = 0; k < 1024; ++k) s = fmaf(bf2f(hr[k]), w[k * 10 + o], s);
  out[t] = s;
}

// ---------- Launch --------------------------------------------------------------
extern "C" void kernel_launch(void* const* d_in, const int* in_sizes, int n_in,
                              void* d_out, int out_size, void* d_ws, size_t ws_size,
                              hipStream_t stream) {
  (void)in_sizes; (void)n_in; (void)out_size; (void)ws_size;
  const float* x    = (const float*)d_in[0];
  const float* W0   = (const float*)d_in[1];
  const float* b0   = (const float*)d_in[2];
  const float* W1   = (const float*)d_in[3];
  const float* b1   = (const float*)d_in[4];
  const float* fcW0 = (const float*)d_in[5];
  const float* fcb0 = (const float*)d_in[6];
  const float* fcW1 = (const float*)d_in[7];
  const float* fcb1 = (const float*)d_in[8];
  float* out = (float*)d_out;

  char* ws = (char*)d_ws;
  size_t off = 0;
  auto wsalloc = [&](size_t bytes) {
    void* p = ws + off;
    off += (bytes + 255) & ~(size_t)255;
    return p;
  };
  __bf16* P0 = (__bf16*)wsalloc(961ULL * 256 * 64 * 2);    // im2col L0
  __bf16* B0 = (__bf16*)wsalloc(961ULL * 64 * 64 * 2);     // W0 bf16, K-padded
  __bf16* H0 = (__bf16*)wsalloc(256ULL * 64 * 961 * 2);    // L0 output (b,c,y,x)
  __bf16* B1 = (__bf16*)wsalloc(196ULL * 128 * 1024 * 2);  // W1 bf16
  __bf16* H1 = (__bf16*)wsalloc(256ULL * 25088 * 2);       // L1 output, FC-flat
  __bf16* W2 = (__bf16*)wsalloc(1024ULL * 25088 * 2);      // fcW0^T bf16
  __bf16* H2 = (__bf16*)wsalloc(256ULL * 1024 * 2);        // FC0 output

  // Prep
  k_cvt_w0<<<(961 * 64 * 64 + 255) / 256, 256, 0, stream>>>(W0, B0);
  k_im2col0<<<(int)((961LL * 256 * 64 + 255) / 256), 256, 0, stream>>>(x, P0);
  k_cvt<<<(int)((196LL * 128 * 1024 + 255) / 256), 256, 0, stream>>>(W1, B1,
                                                                     196LL * 128 * 1024);
  k_cvt_fcw0T<<<(int)((1024LL * 25088 + 255) / 256), 256, 0, stream>>>(fcW0, W2);

  // Layer 0: 961 batched (256x64x64) GEMMs; H0[b*61504 + n*961 + p]
  gemm_bias_act<64, 64, true, __bf16><<<dim3(4, 1, 961), 256, 0, stream>>>(
      P0, 256LL * 64, 64, B0, 64LL * 64, b0, 64,
      H0, 1LL, 61504LL, 961LL, 64);

  // Layer 1: 196 batched (256x128x1024) GEMMs with fused im2col from H0
  gemm_local1<<<dim3(4, 1, 196), 256, 0, stream>>>(H0, B1, b1, H1);

  // FC0: 256x1024x25088 GEMM + relu
  gemm_bias_act<64, 128, true, __bf16><<<dim3(4, 8, 1), 256, 0, stream>>>(
      H1, 0LL, 25088, W2, 0LL, fcb0, 0,
      H2, 0LL, 1024LL, 1LL, 25088);

  // FC1: tiny
  k_fc1<<<10, 256, 0, stream>>>(H2, fcW1, fcb1, out);
}
